// CTRGCNModel_19267223290308
// MI455X (gfx1250) — compile-verified, matrix operand order checked by reading, not run on hardware
//
#include <hip/hip_runtime.h>
#include <hip/hip_bf16.h>
#include <math.h>

// ---------------------------------------------------------------------------
// Types for CDNA5 WMMA
// ---------------------------------------------------------------------------
typedef __attribute__((ext_vector_type(16))) __bf16 v16bf;
typedef __attribute__((ext_vector_type(8)))  __bf16 v8bf;
typedef __attribute__((ext_vector_type(8)))  float  v8f;
typedef __attribute__((ext_vector_type(4)))  int    v4i;

static inline int cdiv(int a, int b) { return (a + b - 1) / b; }
static inline int pad32(int k) { return (k + 31) & ~31; }

// ---------------------------------------------------------------------------
// Async Global->LDS copy (CDNA5 GLOBAL_LOAD_ASYNC_TO_LDS_B128, ASYNCcnt),
// with a synchronous fallback when the builtin is unavailable.
// Builtin signature (from hipcc diagnostic): param 1 is v4i* in the device
// (global, addrspace(1)) address space; param 2 is the LDS destination.
// ---------------------------------------------------------------------------
#if __has_builtin(__builtin_amdgcn_global_load_async_to_lds_b128)
#define CDNA5_HAVE_ASYNC 1
#else
#define CDNA5_HAVE_ASYNC 0
#endif

typedef __attribute__((address_space(1))) v4i as1_v4i;
typedef __attribute__((address_space(3))) v4i as3_v4i;

__device__ __forceinline__ void cp16_g2l(const __bf16* g, __bf16* l)
{
#if CDNA5_HAVE_ASYNC
    __builtin_amdgcn_global_load_async_to_lds_b128(
        (as1_v4i*)(v4i*)(void*)const_cast<__bf16*>(g),
        (as3_v4i*)(v4i*)(void*)l, 0, 0);
#else
    *(v8bf*)l = *(const v8bf*)g;
#endif
}

__device__ __forceinline__ void cp_wait()
{
#if CDNA5_HAVE_ASYNC
#if __has_builtin(__builtin_amdgcn_s_wait_asynccnt)
    __builtin_amdgcn_s_wait_asynccnt(0);
#else
    asm volatile("s_wait_asynccnt 0" ::: "memory");
#endif
#endif
}

// ---------------------------------------------------------------------------
// WMMA GEMM:  Y = epilogue( A[M,Kp](bf16) * B[N,Kp]^T(bf16) )
// 128x64 tile, 8 wave32 waves, 2x2 16x16x32 bf16 fragments per wave,
// double-buffered LDS with async Global->LDS staging.
// Output written channel-major: idx = ((s*Ctot + c_off + n)*TVout + p),
// with m = s*TVout + p  (s = sample, p = position within (T,V) plane).
// epilogue: v = v*scale[n] + shift[n] (+ addend) (ReLU optional)
//   addend_mode: 0 none, 1 full-index residual, 2 per-p broadcast (PA matrix)
// ---------------------------------------------------------------------------
__global__ __launch_bounds__(256)
void k_gemm_wmma(const __bf16* __restrict__ A, const __bf16* __restrict__ B,
                 float* __restrict__ Y,
                 const float* __restrict__ scale, const float* __restrict__ shift,
                 const float* __restrict__ addend, int addend_mode,
                 int M, int N, int Kp, int TVout, int Ctot, int c_off, int relu)
{
    __shared__ __bf16 lA[2][128][32];
    __shared__ __bf16 lB[2][64][32];

    const int tid  = threadIdx.x;
    const int lane = tid & 31;
    const int wave = tid >> 5;
    const int wm = wave & 3;          // 4 waves along M
    const int wn = wave >> 2;         // 2 waves along N
    const int hi = lane >> 4;         // half-wave selector (K-interleave)
    const int lo = lane & 15;
    const int m0 = blockIdx.x * 128;
    const int n0 = blockIdx.y * 64;

    // ---- static chunk assignment for staging ----
    // A: 128 rows x 32 cols; thread covers row (tid>>1), cols [akc, akc+16)
    const int arow = tid >> 1;
    const int akc  = (tid & 1) * 16;
    const bool a_ok = (m0 + arow) < M;
    // B: 64 rows x 32 cols; thread covers row (tid>>2), cols [bkc, bkc+8)
    const int brow = tid >> 2;
    const int bkc  = (tid & 3) * 8;
    const bool b_ok = (n0 + brow) < N;

    // Zero-fill tail rows once: async loads are predicated off for OOB lanes,
    // so these zeros persist across all pipeline stages.
    if (!a_ok) {
        v8bf z = {};
        *(v8bf*)&lA[0][arow][akc] = z; *(v8bf*)&lA[0][arow][akc + 8] = z;
        *(v8bf*)&lA[1][arow][akc] = z; *(v8bf*)&lA[1][arow][akc + 8] = z;
    }
    if (!b_ok) {
        v8bf z = {};
        *(v8bf*)&lB[0][brow][bkc] = z;
        *(v8bf*)&lB[1][brow][bkc] = z;
    }

    const __bf16* agp = A + (size_t)(m0 + arow) * Kp + akc;
    const __bf16* bgp = B + (size_t)(n0 + brow) * Kp + bkc;

    v8f acc[2][2] = {};

    // ---- prologue: stage first K tile into buffer 0 ----
    if (a_ok) { cp16_g2l(agp, &lA[0][arow][akc]); cp16_g2l(agp + 8, &lA[0][arow][akc + 8]); }
    if (b_ok) { cp16_g2l(bgp, &lB[0][brow][bkc]); }
    cp_wait();
    __syncthreads();

    int buf = 0;
    for (int k0 = 0; k0 < Kp; k0 += 32) {
        // ---- stage next K tile into the other buffer (async, no wait) ----
        if (k0 + 32 < Kp) {
            int nb = buf ^ 1;
            if (a_ok) {
                const __bf16* g = agp + k0 + 32;
                cp16_g2l(g, &lA[nb][arow][akc]);
                cp16_g2l(g + 8, &lA[nb][arow][akc + 8]);
            }
            if (b_ok) cp16_g2l(bgp + k0 + 32, &lB[nb][brow][bkc]);
        }

        // ---- 2x2 WMMA fragments per wave on current buffer.
        // 16-bit A/B fragment striping (ISA 7.12.2): lane L holds row (L&15);
        // lanes 0-15 carry K = [0..7]+[16..23], lanes 16-31 K = [8..15]+[24..31].
#pragma unroll
        for (int mi = 0; mi < 2; ++mi) {
            int ar = wm * 32 + mi * 16 + lo;
            v8bf a_lo = *(const v8bf*)&lA[buf][ar][hi * 8];
            v8bf a_hi = *(const v8bf*)&lA[buf][ar][16 + hi * 8];
            v16bf afrag = __builtin_shufflevector(a_lo, a_hi,
                0,1,2,3,4,5,6,7,8,9,10,11,12,13,14,15);
#pragma unroll
            for (int ni = 0; ni < 2; ++ni) {
                int brw = wn * 32 + ni * 16 + lo;
                v8bf b_lo = *(const v8bf*)&lB[buf][brw][hi * 8];
                v8bf b_hi = *(const v8bf*)&lB[buf][brw][16 + hi * 8];
                v16bf bfrag = __builtin_shufflevector(b_lo, b_hi,
                    0,1,2,3,4,5,6,7,8,9,10,11,12,13,14,15);
                acc[mi][ni] = __builtin_amdgcn_wmma_f32_16x16x32_bf16(
                    false, afrag, false, bfrag, (short)0, acc[mi][ni], false, false);
            }
        }

        // all of this wave's async writes for the next buffer must land
        // before any wave crosses the barrier and reads it
        cp_wait();
        __syncthreads();
        buf ^= 1;
    }

    // ---- epilogue: C/D layout (ISA): lane&15 = N col; VGPR r -> M = r + 8*(lane>>4)
#pragma unroll
    for (int mi = 0; mi < 2; ++mi)
#pragma unroll
        for (int ni = 0; ni < 2; ++ni)
#pragma unroll
            for (int r = 0; r < 8; ++r) {
                int m = m0 + wm * 32 + mi * 16 + hi * 8 + r;
                int n = n0 + wn * 32 + ni * 16 + lo;
                if (m < M && n < N) {
                    float v = acc[mi][ni][r];
                    v = v * scale[n] + shift[n];
                    int s = m / TVout, p = m - s * TVout;
                    size_t idx = ((size_t)s * Ctot + c_off + n) * (size_t)TVout + p;
                    if (addend_mode == 1)      v += addend[idx];
                    else if (addend_mode == 2) v += addend[p];
                    if (relu) v = fmaxf(v, 0.0f);
                    Y[idx] = v;
                }
            }
}

// ---------------------------------------------------------------------------
// Helper kernels
// ---------------------------------------------------------------------------

// (NM,C,Tin,V) fp32 channel-major -> [M = NM*Tout*V, Kp] bf16 row-major, temporal stride
__global__ void k_act_to_bf16(const float* __restrict__ x, __bf16* __restrict__ out,
                              int NM, int C, int Tin, int V, int stride, int Kp, int Tout)
{
    size_t idx = (size_t)blockIdx.x * blockDim.x + threadIdx.x;
    size_t tot = (size_t)NM * Tout * V * Kp;
    if (idx >= tot) return;
    int k = (int)(idx % Kp);
    size_t m = idx / Kp;
    int p = (int)(m % ((size_t)Tout * V));
    int s = (int)(m / ((size_t)Tout * V));
    int t = p / V, v = p - t * V;
    float val = 0.0f;
    if (k < C) val = x[(((size_t)s * C + k) * Tin + (size_t)t * stride) * V + v];
    out[idx] = (__bf16)val;
}

// weights (N,C) fp32 -> bf16 [N, Kp] zero-padded
__global__ void k_w_bf16(const float* __restrict__ w, __bf16* __restrict__ out,
                         int N, int C, int Kp)
{
    int idx = blockIdx.x * blockDim.x + threadIdx.x;
    if (idx >= N * Kp) return;
    int k = idx % Kp, n = idx / Kp;
    out[idx] = (__bf16)(k < C ? w[(size_t)n * C + k] : 0.0f);
}

// per-channel epilogue (scale, shift)
// mode 0: scale=1, shift=bias          (plain conv bias)
// mode 1: BN folding, optional conv bias: s=g/sqrt(v+eps), sh=b-m*s(+bias*s)
// mode 2: attention: scale=alpha, shift=bias*alpha
__global__ void k_prep(float* scale, float* shift, int C, int mode,
                       const float* g, const float* bb, const float* mm, const float* vv,
                       const float* bias, const float* alpha)
{
    int c = blockIdx.x * blockDim.x + threadIdx.x;
    if (c >= C) return;
    if (mode == 0) {
        scale[c] = 1.0f; shift[c] = bias ? bias[c] : 0.0f;
    } else if (mode == 1) {
        float s = g[c] * rsqrtf(vv[c] + 1e-5f);
        scale[c] = s;
        shift[c] = bb[c] - mm[c] * s + (bias ? bias[c] * s : 0.0f);
    } else {
        float al = alpha[0];
        scale[c] = al; shift[c] = bias[c] * al;
    }
}

// data_bn: channels indexed (v*C + c) over input layout (N,C,T,V,1)
__global__ void k_data_bn(const float* __restrict__ x, float* __restrict__ out,
                          const float* g, const float* b, const float* m, const float* vv,
                          int NM, int C, int T, int V)
{
    size_t idx = (size_t)blockIdx.x * blockDim.x + threadIdx.x;
    size_t tot = (size_t)NM * C * T * V;
    if (idx >= tot) return;
    int v  = (int)(idx % V);
    int c  = (int)((idx / ((size_t)V * T)) % C);
    int ch = v * C + c;
    float s = g[ch] * rsqrtf(vv[ch] + 1e-5f);
    out[idx] = x[idx] * s + (b[ch] - m[ch] * s);
}

// mean over T: (NM,C,T,V) -> (NM,C,V)
__global__ void k_mean_T(const float* __restrict__ x, float* __restrict__ out,
                         int NM, int C, int T, int V)
{
    int idx = blockIdx.x * blockDim.x + threadIdx.x;
    if (idx >= NM * C * V) return;
    int v = idx % V;
    int c = (idx / V) % C;
    int n = idx / (V * C);
    const float* p = x + ((size_t)n * C + c) * T * V + v;
    float s = 0.0f;
    for (int t = 0; t < T; ++t) s += p[(size_t)t * V];
    out[((size_t)n * C + c) * V + v] = s / (float)T;
}

// d[n,r,u,v] = tanh(x1[n,r,u] - x2[n,r,v])
__global__ void k_tanh_diff(const float* __restrict__ x1, const float* __restrict__ x2,
                            float* __restrict__ d, int NM, int R, int V)
{
    size_t idx = (size_t)blockIdx.x * blockDim.x + threadIdx.x;
    size_t tot = (size_t)NM * R * V * V;
    if (idx >= tot) return;
    int v = (int)(idx % V);
    int u = (int)((idx / V) % V);
    size_t nr = idx / ((size_t)V * V);   // n*R + r
    d[idx] = tanhf(x1[nr * V + u] - x2[nr * V + v]);
}

// y[n,c,t,u] (+)= sum_v a[n,c,u,v] * x3[n,c,t,v]
__global__ void k_apply_attn(const float* __restrict__ a, const float* __restrict__ x3,
                             float* __restrict__ y, int NM, int C, int T, int V, int accum)
{
    size_t idx = (size_t)blockIdx.x * blockDim.x + threadIdx.x;
    size_t tot = (size_t)NM * C * T * V;
    if (idx >= tot) return;
    int u = (int)(idx % V);
    int t = (int)((idx / V) % T);
    size_t nc = idx / ((size_t)V * T);   // n*C + c
    const float* ar = a  + (nc * V + u) * V;
    const float* xr = x3 + (nc * T + t) * V;
    float s = 0.0f;
    for (int v = 0; v < V; ++v) s += ar[v] * xr[v];
    if (accum) y[idx] += s; else y[idx] = s;
}

// out = [relu]( x (*scale[c]+shift[c] if scale) (+ addend) )
__global__ void k_bn_add_relu(const float* __restrict__ x,
                              const float* __restrict__ scale, const float* __restrict__ shift,
                              const float* __restrict__ addend, float* __restrict__ out,
                              int NM, int C, int TV, int relu)
{
    size_t idx = (size_t)blockIdx.x * blockDim.x + threadIdx.x;
    size_t tot = (size_t)NM * C * TV;
    if (idx >= tot) return;
    int c = (int)((idx / TV) % C);
    float v = x[idx];
    if (scale)  v = v * scale[c] + shift[c];
    if (addend) v += addend[idx];
    if (relu)   v = fmaxf(v, 0.0f);
    out[idx] = v;
}

// 5x1 dilated temporal conv + fused BN (scale/shift include conv bias)
__global__ void k_tconv5(const float* __restrict__ x, const float* __restrict__ w,
                         const float* __restrict__ scale, const float* __restrict__ shift,
                         float* __restrict__ y, int NM, int Cb, int Tin, int V,
                         int stride, int dil, int Tout, int Ctot, int c_off)
{
    size_t idx = (size_t)blockIdx.x * blockDim.x + threadIdx.x;
    size_t tot = (size_t)NM * Cb * Tout * V;
    if (idx >= tot) return;
    int v  = (int)(idx % V);
    int to = (int)((idx / V) % Tout);
    int o  = (int)((idx / ((size_t)V * Tout)) % Cb);
    int n  = (int)(idx / ((size_t)V * Tout * Cb));
    float acc = 0.0f;
    for (int c = 0; c < Cb; ++c) {
        const float* xr = x + ((size_t)n * Cb + c) * Tin * V + v;
        const float* wr = w + ((size_t)o * Cb + c) * 5;
#pragma unroll
        for (int j = 0; j < 5; ++j) {
            int ti = to * stride + (j - 2) * dil;
            if (ti >= 0 && ti < Tin) acc += wr[j] * xr[(size_t)ti * V];
        }
    }
    y[(((size_t)n * Ctot + c_off + o) * Tout + to) * V + v] = acc * scale[o] + shift[o];
}

// maxpool window 3, pad 1, temporal stride s, fused BN
__global__ void k_maxpool_bn(const float* __restrict__ x,
                             const float* __restrict__ scale, const float* __restrict__ shift,
                             float* __restrict__ y, int NM, int Cb, int Tin, int V,
                             int stride, int Tout, int Ctot, int c_off)
{
    size_t idx = (size_t)blockIdx.x * blockDim.x + threadIdx.x;
    size_t tot = (size_t)NM * Cb * Tout * V;
    if (idx >= tot) return;
    int v  = (int)(idx % V);
    int to = (int)((idx / V) % Tout);
    int c  = (int)((idx / ((size_t)V * Tout)) % Cb);
    int n  = (int)(idx / ((size_t)V * Tout * Cb));
    float m = -3.402823466e38f;
    for (int j = -1; j <= 1; ++j) {
        int ti = to * stride + j;
        if (ti >= 0 && ti < Tin)
            m = fmaxf(m, x[(((size_t)n * Cb + c) * Tin + ti) * V + v]);
    }
    y[(((size_t)n * Ctot + c_off + c) * Tout + to) * V + v] = m * scale[c] + shift[c];
}

// global average over (T*V) then 256->2 FC; one block per sample
__global__ void k_final(const float* __restrict__ z, const float* __restrict__ fw,
                        const float* __restrict__ fb, float* __restrict__ out,
                        int C, int TV)
{
    __shared__ float sm[256];
    int n = blockIdx.x, c = threadIdx.x;
    float s = 0.0f;
    const float* p = z + ((size_t)n * C + c) * TV;
    for (int i = 0; i < TV; ++i) s += p[i];
    sm[c] = s / (float)TV;
    __syncthreads();
    if (c < 2) {
        float acc = fb[c];
        for (int k = 0; k < C; ++k) acc += fw[(size_t)c * C + k] * sm[k];
        out[n * 2 + c] = acc;
    }
}

// ---------------------------------------------------------------------------
// Parameter extraction (mirrors jax tree flatten: dicts by sorted key)
// ---------------------------------------------------------------------------
struct Cursor { void* const* in; int i; const float* next() { return (const float*)in[i++]; } };
struct BN4 { const float *b, *g, *m, *v; };
struct TCONVP { const float* b; BN4 bn; const float* w; };
struct CT { const float *b1,*b2,*b3,*b4,*w1,*w2,*w3,*w4; };
struct GCNP { const float *PA, *alpha; BN4 bn; CT ct[3]; int has_down;
              const float* down_b; BN4 down_bn; const float* down_w; };
struct BranchP { const float* b; BN4 bn; TCONVP tc; BN4 bn2; const float* w; };
struct BlockP { GCNP gcn; int has_res; TCONVP res; BranchP br[4]; };

static BN4 getBN(Cursor& c) { BN4 r; r.b=c.next(); r.g=c.next(); r.m=c.next(); r.v=c.next(); return r; }
static TCONVP getTC(Cursor& c) { TCONVP t; t.b=c.next(); t.bn=getBN(c); t.w=c.next(); return t; }
static CT getCT(Cursor& c) { CT t;
    t.b1=c.next(); t.b2=c.next(); t.b3=c.next(); t.b4=c.next();
    t.w1=c.next(); t.w2=c.next(); t.w3=c.next(); t.w4=c.next(); return t; }

static BlockP getBlock(Cursor& c, int cin, int cout, int residual) {
    BlockP B{};
    B.gcn.PA = c.next(); B.gcn.alpha = c.next(); B.gcn.bn = getBN(c);
    for (int i = 0; i < 3; ++i) B.gcn.ct[i] = getCT(c);
    if (cin != cout) { B.gcn.has_down = 1; B.gcn.down_b = c.next();
                       B.gcn.down_bn = getBN(c); B.gcn.down_w = c.next(); }
    if (residual && cin != cout) { B.has_res = 1; B.res = getTC(c); }
    for (int i = 0; i < 2; ++i) { B.br[i].b = c.next(); B.br[i].bn = getBN(c);
                                  B.br[i].tc = getTC(c); B.br[i].w = c.next(); }
    B.br[2].b = c.next(); B.br[2].bn = getBN(c); B.br[2].bn2 = getBN(c); B.br[2].w = c.next();
    B.br[3].b = c.next(); B.br[3].bn = getBN(c); B.br[3].w = c.next();
    return B;
}

struct Cfg { int cin, cout, T, stride, residual; };
static const Cfg g_cfg[10] = {
    {3,64,256,1,0},{64,64,256,1,1},{64,64,256,1,1},{64,64,256,1,1},
    {64,128,256,2,1},{128,128,128,1,1},{128,128,128,1,1},
    {128,256,128,2,1},{256,256,64,1,1},{256,256,64,1,1}};

// ---------------------------------------------------------------------------
extern "C" void kernel_launch(void* const* d_in, const int* in_sizes, int n_in,
                              void* d_out, int out_size, void* d_ws, size_t ws_size,
                              hipStream_t stream)
{
    (void)in_sizes; (void)n_in; (void)out_size; (void)ws_size;
    const int NM = 64, V = 17;

    const float* xin0 = (const float*)d_in[0];
    Cursor cu{d_in, 1};
    BN4 data_bn = getBN(cu);
    const float* fc_b = cu.next();
    const float* fc_w = cu.next();
    BlockP blk[10];
    static const int sorted_order[10] = {0, 9, 1, 2, 3, 4, 5, 6, 7, 8}; // l1,l10,l2..l9
    for (int oi = 0; oi < 10; ++oi) {
        int bi = sorted_order[oi];
        blk[bi] = getBlock(cu, g_cfg[bi].cin, g_cfg[bi].cout, g_cfg[bi].residual);
    }

    // ---- workspace bump allocator (reset every call: deterministic) ----
    char* wsp = (char*)d_ws; size_t off = 0;
    auto allocF = [&](size_t n) { float* p = (float*)(wsp + off);
                                  off = (off + n * 4 + 255) & ~(size_t)255; return p; };
    auto allocH = [&](size_t n) { __bf16* p = (__bf16*)(wsp + off);
                                  off = (off + n * 2 + 255) & ~(size_t)255; return p; };

    const size_t ACT = (size_t)NM * 16384 * V;   // C*T == 16384 for all post-l1 layers
    float* act0 = allocF(ACT);
    float* act1 = allocF(ACT);
    float* act2 = allocF(ACT);
    float* x3b  = allocF(ACT);
    float* ybuf = allocF(ACT);
    float* rbuf = allocF(ACT);
    float* zbuf = allocF(ACT / 2);
    float* xm   = allocF((size_t)NM * 256 * V);
    float* x1m  = allocF((size_t)NM * 32 * V);
    float* x2m  = allocF((size_t)NM * 32 * V);
    float* dbuf = allocF((size_t)NM * 32 * V * V);
    float* abuf = allocF((size_t)NM * 256 * V * V);
    __bf16* Xbf = allocH(ACT);
    __bf16* Gbf = allocH(ACT);
    __bf16* Sbf = allocH(ACT);
    __bf16* Mbf = allocH((size_t)NM * V * 256);
    __bf16* Dbf = allocH((size_t)NM * V * V * 32);
    __bf16* Wbf = allocH((size_t)256 * 256);
    float* scl = allocF(256);
    float* shf = allocF(256);

    auto ew = [&](size_t tot) { return (int)((tot + 255) / 256); };

    // GEMM wrapper: converts weights, prepares epilogue, launches WMMA kernel
    auto gemm = [&](const __bf16* A, const float* Wf, int M, int N, int K,
                    int epi_mode, const BN4* bn, const float* bias, const float* alpha,
                    const float* addend, int addend_mode,
                    float* Y, int TVout, int Ctot, int c_off, int relu) {
        int Kp = pad32(K);
        k_w_bf16<<<cdiv(N * Kp, 256), 256, 0, stream>>>(Wf, Wbf, N, K, Kp);
        k_prep<<<cdiv(N, 64), 64, 0, stream>>>(scl, shf, N, epi_mode,
            bn ? bn->g : nullptr, bn ? bn->b : nullptr,
            bn ? bn->m : nullptr, bn ? bn->v : nullptr, bias, alpha);
        dim3 g(cdiv(M, 128), cdiv(N, 64));
        k_gemm_wmma<<<g, 256, 0, stream>>>(A, Wbf, Y, scl, shf, addend, addend_mode,
                                           M, N, Kp, TVout, Ctot, c_off, relu);
    };
    auto actbf = [&](const float* src, __bf16* dst, int C, int Tin, int stride) {
        int Kp = pad32(C); int Tout = (Tin + stride - 1) / stride;
        size_t tot = (size_t)NM * Tout * V * Kp;
        k_act_to_bf16<<<ew(tot), 256, 0, stream>>>(src, dst, NM, C, Tin, V, stride, Kp, Tout);
    };

    // ---- stage 0: data BN ----
    k_data_bn<<<ew((size_t)NM * 3 * 256 * V), 256, 0, stream>>>(
        xin0, act0, data_bn.g, data_bn.b, data_bn.m, data_bn.v, NM, 3, 256, V);

    float* pin = act0; float* pg = act1; float* pout = act2;

    for (int bi = 0; bi < 10; ++bi) {
        const Cfg& c = g_cfg[bi]; BlockP& P = blk[bi];
        const int Cin = c.cin, Cout = c.cout, Tin = c.T, S = c.stride;
        const int Tout = Tin / S;
        const int R = (Cin == 3) ? 8 : Cin / 8;
        const int Mfull = NM * Tin * V, Mout = NM * Tout * V;

        // input to bf16 (reused by all 3 CTRGCs + down)
        actbf(pin, Xbf, Cin, Tin, 1);
        // temporal mean path: mean_T then 1x1 conv == conv then mean_T (exact)
        k_mean_T<<<ew((size_t)NM * Cin * V), 256, 0, stream>>>(pin, xm, NM, Cin, Tin, V);
        k_act_to_bf16<<<ew((size_t)NM * V * pad32(Cin)), 256, 0, stream>>>(
            xm, Mbf, NM, Cin, 1, V, 1, pad32(Cin), 1);

        // ---- GCN: 3 CTRGC branches summed into ybuf ----
        for (int i = 0; i < 3; ++i) {
            CT& q = P.gcn.ct[i];
            gemm(Mbf, q.w1, NM * V, R, Cin, 0, nullptr, q.b1, nullptr,
                 nullptr, 0, x1m, V, R, 0, 0);
            gemm(Mbf, q.w2, NM * V, R, Cin, 0, nullptr, q.b2, nullptr,
                 nullptr, 0, x2m, V, R, 0, 0);
            gemm(Xbf, q.w3, Mfull, Cout, Cin, 0, nullptr, q.b3, nullptr,
                 nullptr, 0, x3b, Tin * V, Cout, 0, 0);
            k_tanh_diff<<<ew((size_t)NM * R * V * V), 256, 0, stream>>>(x1m, x2m, dbuf, NM, R, V);
            k_act_to_bf16<<<ew((size_t)NM * V * V * pad32(R)), 256, 0, stream>>>(
                dbuf, Dbf, NM, R, V, V, 1, pad32(R), V);
            // a = alpha*(w4 . d + b4) + PA[i]  (WMMA GEMM, per-(u,v) PA addend)
            gemm(Dbf, q.w4, NM * V * V, Cout, R, 2, nullptr, q.b4, P.gcn.alpha,
                 P.gcn.PA + (size_t)i * V * V, 2, abuf, V * V, Cout, 0, 0);
            k_apply_attn<<<ew((size_t)NM * Cout * Tin * V), 256, 0, stream>>>(
                abuf, x3b, ybuf, NM, Cout, Tin, V, i > 0);
        }
        // residual for GCN (down-projection or identity)
        const float* gres;
        if (P.gcn.has_down) {
            gemm(Xbf, P.gcn.down_w, Mfull, Cout, Cin, 1, &P.gcn.down_bn, P.gcn.down_b,
                 nullptr, nullptr, 0, rbuf, Tin * V, Cout, 0, 0);
            gres = rbuf;
        } else gres = pin;
        k_prep<<<cdiv(Cout, 64), 64, 0, stream>>>(scl, shf, Cout, 1,
            P.gcn.bn.g, P.gcn.bn.b, P.gcn.bn.m, P.gcn.bn.v, nullptr, nullptr);
        k_bn_add_relu<<<ew((size_t)NM * Cout * Tin * V), 256, 0, stream>>>(
            ybuf, scl, shf, gres, pg, NM, Cout, Tin * V, 1);

        // ---- TCN: 4 branches concatenated into pout ----
        actbf(pg, Gbf, Cout, Tin, 1);
        const int bc = Cout / 4;
        for (int i = 0; i < 2; ++i) {               // dilated 5x1 branches (d=1,2)
            BranchP& br = P.br[i];
            gemm(Gbf, br.w, Mfull, bc, Cout, 1, &br.bn, br.b, nullptr,
                 nullptr, 0, zbuf, Tin * V, bc, 0, 1);
            k_prep<<<cdiv(bc, 64), 64, 0, stream>>>(scl, shf, bc, 1,
                br.tc.bn.g, br.tc.bn.b, br.tc.bn.m, br.tc.bn.v, br.tc.b, nullptr);
            k_tconv5<<<ew((size_t)NM * bc * Tout * V), 256, 0, stream>>>(
                zbuf, br.tc.w, scl, shf, pout, NM, bc, Tin, V, S, i + 1, Tout, Cout, i * bc);
        }
        {                                            // maxpool branch
            BranchP& br = P.br[2];
            gemm(Gbf, br.w, Mfull, bc, Cout, 1, &br.bn, br.b, nullptr,
                 nullptr, 0, zbuf, Tin * V, bc, 0, 1);
            k_prep<<<cdiv(bc, 64), 64, 0, stream>>>(scl, shf, bc, 1,
                br.bn2.g, br.bn2.b, br.bn2.m, br.bn2.v, nullptr, nullptr);
            k_maxpool_bn<<<ew((size_t)NM * bc * Tout * V), 256, 0, stream>>>(
                zbuf, scl, shf, pout, NM, bc, Tin, V, S, Tout, Cout, 2 * bc);
        }
        {                                            // strided 1x1 branch
            BranchP& br = P.br[3];
            actbf(pg, Sbf, Cout, Tin, S);
            gemm(Sbf, br.w, Mout, bc, Cout, 1, &br.bn, br.b, nullptr,
                 nullptr, 0, pout, Tout * V, Cout, 3 * bc, 0);
        }

        // ---- block residual + ReLU ----
        const float* radd = nullptr;
        if (c.residual) {
            if (P.has_res) {                         // 1x1 strided conv + BN on block input
                actbf(pin, Sbf, Cin, Tin, S);
                gemm(Sbf, P.res.w, Mout, Cout, Cin, 1, &P.res.bn, P.res.b, nullptr,
                     nullptr, 0, rbuf, Tout * V, Cout, 0, 0);
                radd = rbuf;
            } else radd = pin;                       // identity (S==1, Cin==Cout)
        }
        k_bn_add_relu<<<ew((size_t)NM * Cout * Tout * V), 256, 0, stream>>>(
            pout, nullptr, nullptr, radd, pout, NM, Cout, Tout * V, 1);

        // rotate activation buffers
        float* t = pin; pin = pout; pout = t;
    }

    // ---- global average pool + FC ----
    k_final<<<NM, 256, 0, stream>>>(pin, fc_w, fc_b, (float*)d_out, 256, 64 * V);
}
